// TFCTRLMainLayer_42554535969284
// MI455X (gfx1250) — compile-verified, matrix operand order checked by reading, not run on hardware
//
#include <hip/hip_runtime.h>

// ---------------- problem constants ----------------
constexpr int BB   = 2;
constexpr int SS   = 1024;
constexpr int DD   = 1280;
constexpr int HH   = 16;
constexpr int DHC  = 80;        // D / H
constexpr int DFFC = 8192;
constexpr int LL   = 4;
constexpr int BSR  = BB * SS;   // 2048 rows

#define DEV __device__ __forceinline__

typedef __bf16 v16bf __attribute__((ext_vector_type(16)));
typedef float  v8f   __attribute__((ext_vector_type(8)));
typedef float  f32x4 __attribute__((ext_vector_type(4)));

struct F4x2 { f32x4 lo, hi; };

DEV v16bf make_frag(f32x4 lo, f32x4 hi) {
    F4x2 t; t.lo = lo; t.hi = hi;
    return __builtin_bit_cast(v16bf, t);
}
DEV f32x4 ld16(const __bf16* p) { return *reinterpret_cast<const f32x4*>(p); }

DEV v8f wmma_bf16(v16bf a, v16bf b, v8f c) {
    return __builtin_amdgcn_wmma_f32_16x16x32_bf16(
        /*neg_a=*/false, a, /*neg_b=*/false, b,
        /*c_mod=*/(short)0, c, /*reuse_a=*/false, /*reuse_b=*/false);
}

// CDNA5 async global->LDS copy (ASYNCcnt-tracked), GVS mode:
//   addr = SGPR base + 32-bit VGPR byte offset; vdst = per-lane LDS byte address.
DEV void async_b128_to_lds(const void* lds_addr, unsigned goff_bytes, const void* sbase) {
    unsigned loff = (unsigned)(size_t)lds_addr;   // LDS aperture: addr[31:0] = LDS offset
    asm volatile("global_load_async_to_lds_b128 %0, %1, %2"
                 :: "v"(loff), "v"(goff_bytes), "s"(sbase)
                 : "memory");
}
DEV void wait_async0() { asm volatile("s_wait_asynccnt 0" ::: "memory"); }

// ---------------- f32 -> bf16 convert ----------------
__global__ __launch_bounds__(256) void cvt_kernel(const float* __restrict__ in,
                                                  __bf16* __restrict__ out, size_t n) {
    size_t i = ((size_t)blockIdx.x * 256 + threadIdx.x) * 4;
    if (i >= n) return;
    f32x4 v = *reinterpret_cast<const f32x4*>(in + i);
    union { unsigned long long u; __bf16 h[4]; } o;
    o.h[0] = (__bf16)v[0]; o.h[1] = (__bf16)v[1];
    o.h[2] = (__bf16)v[2]; o.h[3] = (__bf16)v[3];
    *reinterpret_cast<unsigned long long*>(out + i) = o.u;
}

// ---------------- embedding + sinusoidal positions ----------------
__global__ __launch_bounds__(256) void embed_kernel(const int* __restrict__ ids,
                                                    const float* __restrict__ emb,
                                                    float* __restrict__ x) {
    const int row = blockIdx.x;            // b*S + s
    const int s   = row % SS;
    const int id  = ids[row];
    const float* er = emb + (size_t)id * DD;
    float* xr = x + (size_t)row * DD;
    const float sqrtd = 35.77708764f;      // sqrt(1280)
    for (int d = threadIdx.x; d < DD; d += 256) {
        float pe;
        if (d < DD / 2) {
            float rate = __powf(10000.0f, -2.0f * (float)d / (float)DD);
            pe = __sinf((float)s * rate);
        } else {
            int j = d - DD / 2;
            float rate = __powf(10000.0f, -2.0f * (float)j / (float)DD);
            pe = __cosf((float)s * rate);
        }
        xr[d] = er[d] * sqrtd + pe;
    }
}

// ---------------- layernorm (row per block), templated output ----------------
template <typename OutT>
__global__ __launch_bounds__(256) void layernorm_kernel(const float* __restrict__ x,
                                                        const float* __restrict__ g,
                                                        const float* __restrict__ beta,
                                                        OutT* __restrict__ out) {
    const int row = blockIdx.x;
    const float* xr = x + (size_t)row * DD;
    const int lane = threadIdx.x & 31, wave = threadIdx.x >> 5;
    __shared__ float red[8];

    float v[5];
    float s = 0.f;
#pragma unroll
    for (int i = 0; i < 5; ++i) { v[i] = xr[threadIdx.x + i * 256]; s += v[i]; }
#pragma unroll
    for (int m = 16; m >= 1; m >>= 1) s += __shfl_xor(s, m, 32);
    if (lane == 0) red[wave] = s;
    __syncthreads();
    float tot = 0.f;
#pragma unroll
    for (int i = 0; i < 8; ++i) tot += red[i];
    const float mu = tot * (1.0f / DD);
    __syncthreads();

    float vs = 0.f;
#pragma unroll
    for (int i = 0; i < 5; ++i) { float d = v[i] - mu; vs += d * d; }
#pragma unroll
    for (int m = 16; m >= 1; m >>= 1) vs += __shfl_xor(vs, m, 32);
    if (lane == 0) red[wave] = vs;
    __syncthreads();
    tot = 0.f;
#pragma unroll
    for (int i = 0; i < 8; ++i) tot += red[i];
    const float rstd = rsqrtf(tot * (1.0f / DD) + 1e-6f);

#pragma unroll
    for (int i = 0; i < 5; ++i) {
        int d = threadIdx.x + i * 256;
        float o = (v[i] - mu) * rstd * g[d] + beta[d];
        out[(size_t)row * DD + d] = (OutT)o;
    }
}

// ---------------- WMMA bf16 GEMM: C[M,N] = A[M,K] @ B[K,N] + bias (+resid, relu) ----
// 128x128x32 tiles, 256 threads = 8 waves (2x4), 4x2 WMMA tiles/wave.
// Double-buffered LDS; A tiles fetched with async global->LDS (ASYNCcnt),
// overlapped with WMMA on the current tile. B tiles transposed via VGPR path.
template <bool RELU, bool RES, typename OutT>
__global__ __launch_bounds__(256) void gemm_kernel(const __bf16* __restrict__ A,
                                                   const __bf16* __restrict__ Bw,
                                                   const float* __restrict__ bias,
                                                   const float* __restrict__ resid,
                                                   OutT* __restrict__ C,
                                                   int M, int N, int K) {
    constexpr int BM = 128, BN = 128, BK = 32, LDT = 40; // 80B rows, 16B aligned
    __shared__ __align__(16) __bf16 As[2][BM * LDT];
    __shared__ __align__(16) __bf16 Bs[2][BN * LDT];     // B stored transposed: [n][k]

    const int tid  = threadIdx.x;
    const int wave = tid >> 5, lane = tid & 31;
    const int wm   = wave >> 2, wn = wave & 3;           // 2 x 4 waves
    const int half = lane >> 4, mr = lane & 15;
    const int m0 = blockIdx.y * BM, n0 = blockIdx.x * BN;

    const v8f zacc = {0.f, 0.f, 0.f, 0.f, 0.f, 0.f, 0.f, 0.f};
    v8f acc[4][2];
#pragma unroll
    for (int i = 0; i < 4; ++i)
#pragma unroll
        for (int j = 0; j < 2; ++j) acc[i][j] = zacc;

    auto load_tiles = [&](int k0, int buf) {
#pragma unroll
        for (int i = 0; i < 2; ++i) {
            int c = tid + i * 256;                       // 512 chunks of 8 bf16
            // A tile [128][32]: async DMA straight into LDS, no VGPR round-trip
            int r = c >> 2, col = (c & 3) << 3;
            unsigned goff = (unsigned)((((size_t)(m0 + r)) * K + k0 + col) * 2);
            async_b128_to_lds(&As[buf][r * LDT + col], goff, A);
            // B tile: read [32][128] rows, store transposed [n][k]
            int kr = c >> 4, nc = (c & 15) << 3;
            union { f32x4 f; __bf16 h[8]; } u;
            u.f = *reinterpret_cast<const f32x4*>(Bw + (size_t)(k0 + kr) * N + n0 + nc);
#pragma unroll
            for (int j = 0; j < 8; ++j) Bs[buf][(nc + j) * LDT + kr] = u.h[j];
        }
    };

    load_tiles(0, 0);
    wait_async0();
    __syncthreads();

    const int nk = K / BK;
    for (int t = 0; t < nk; ++t) {
        const int buf = t & 1;
        if (t + 1 < nk) load_tiles((t + 1) * BK, buf ^ 1);   // prefetch next tile

        v16bf af[4], bf[2];
#pragma unroll
        for (int u = 0; u < 4; ++u) {
            const __bf16* p = &As[buf][(wm * 64 + u * 16 + mr) * LDT + half * 8];
            af[u] = make_frag(ld16(p), ld16(p + 16));
        }
#pragma unroll
        for (int u = 0; u < 2; ++u) {
            const __bf16* p = &Bs[buf][(wn * 32 + u * 16 + mr) * LDT + half * 8];
            bf[u] = make_frag(ld16(p), ld16(p + 16));
        }
#pragma unroll
        for (int i = 0; i < 4; ++i)
#pragma unroll
            for (int j = 0; j < 2; ++j)
                acc[i][j] = wmma_bf16(af[i], bf[j], acc[i][j]);

        wait_async0();       // this wave's async writes into buf^1 have landed
        __syncthreads();     // everyone done reading buf, writes visible
    }

#pragma unroll
    for (int j = 0; j < 2; ++j) {
        const int n = n0 + wn * 32 + j * 16 + mr;
        const float bb = bias[n];
#pragma unroll
        for (int i = 0; i < 4; ++i) {
#pragma unroll
            for (int r = 0; r < 8; ++r) {
                const int m = m0 + wm * 64 + i * 16 + half * 8 + r;
                float v = acc[i][j][r] + bb;
                if constexpr (RES) v += resid[(size_t)m * N + n];
                if constexpr (RELU) v = fmaxf(v, 0.f);
                C[(size_t)m * N + n] = (OutT)v;
            }
        }
    }
}

// ---------------- flash attention: one (b,h), 128 queries / block, 8 waves ----------
__global__ __launch_bounds__(256) void attention_kernel(const __bf16* __restrict__ q,
                                                        const __bf16* __restrict__ k,
                                                        const __bf16* __restrict__ v,
                                                        __bf16* __restrict__ ctx) {
    constexpr float scale = 0.11180339887f;        // 1/sqrt(80)
    __shared__ __align__(16) __bf16 Ks[32 * 80];   // [key][dh]
    __shared__ __align__(16) __bf16 Vs[80 * 32];   // [dh][key] (transposed)
    __shared__ __align__(16) __bf16 Ps[8][512];    // per-wave probs 16x32

    const int bh = blockIdx.y;
    const int b = bh / HH, h = bh % HH;
    const int q0 = blockIdx.x * 128;
    const int tid = threadIdx.x, wave = tid >> 5, lane = tid & 31;
    const int half = lane >> 4, mr = lane & 15;

    const size_t baseQ = ((size_t)b * SS) * DD + (size_t)h * DHC;
    const __bf16* qp = q + baseQ;
    const __bf16* kp = k + baseQ;
    const __bf16* vp = v + baseQ;

    const int qw = q0 + wave * 16;
    const f32x4 zero4 = {0.f, 0.f, 0.f, 0.f};
    const v8f zacc = {0.f, 0.f, 0.f, 0.f, 0.f, 0.f, 0.f, 0.f};

    // Q fragments: 3 K-chunks of 32 (dh 0..79, pad 80..95 with zero)
    v16bf qf[3];
    {
        const __bf16* qr = qp + (size_t)(qw + mr) * DD;
#pragma unroll
        for (int c = 0; c < 3; ++c) {
            f32x4 lo = ld16(qr + c * 32 + half * 8);
            f32x4 hi = (c < 2) ? ld16(qr + c * 32 + half * 8 + 16) : zero4;
            qf[c] = make_frag(lo, hi);
        }
    }

    v8f cacc[5];
#pragma unroll
    for (int t = 0; t < 5; ++t) cacc[t] = zacc;
    float mst[8], lst[8];
#pragma unroll
    for (int r = 0; r < 8; ++r) { mst[r] = -1e30f; lst[r] = 0.f; }

    const int ktiles = (q0 + 128) / 32;            // causal: keys up to block max
    for (int kt = 0; kt < ktiles; ++kt) {
        const int key0 = kt * 32;
        __syncthreads();
        // K tile: async global->LDS b128 (row-major copy, no transpose needed)
        for (int i = tid; i < 320; i += 256) {     // 32*80/8 chunks
            int r = i / 10, c8 = (i % 10) * 8;
            unsigned goff = (unsigned)((((size_t)(key0 + r)) * DD + c8) * 2);
            async_b128_to_lds(&Ks[r * 80 + c8], goff, kp);
        }
        // V tile: transposed via VGPR path
        for (int i = tid; i < 32 * 80; i += 256) {
            int r = i / 80, c2 = i % 80;
            Vs[c2 * 32 + r] = vp[(size_t)(key0 + r) * DD + c2];
        }
        wait_async0();
        __syncthreads();

        if (qw + 15 >= key0) {
            v8f s0 = zacc, s1 = zacc;
#pragma unroll
            for (int c = 0; c < 3; ++c) {
                const __bf16* kr0 = &Ks[(mr) * 80 + c * 32 + half * 8];
                const __bf16* kr1 = &Ks[(16 + mr) * 80 + c * 32 + half * 8];
                f32x4 hi0 = (c < 2) ? ld16(kr0 + 16) : zero4;
                f32x4 hi1 = (c < 2) ? ld16(kr1 + 16) : zero4;
                s0 = wmma_bf16(qf[c], make_frag(ld16(kr0), hi0), s0);
                s1 = wmma_bf16(qf[c], make_frag(ld16(kr1), hi1), s1);
            }
            const int rowbase = qw + half * 8;
            const int kn0 = key0 + mr, kn1 = key0 + 16 + mr;
            float lg0[8], lg1[8];
#pragma unroll
            for (int r = 0; r < 8; ++r) {
                lg0[r] = s0[r] * scale + ((kn0 > rowbase + r) ? -10000.f : 0.f);
                lg1[r] = s1[r] * scale + ((kn1 > rowbase + r) ? -10000.f : 0.f);
            }
#pragma unroll
            for (int r = 0; r < 8; ++r) {
                float mx = fmaxf(lg0[r], lg1[r]);
#pragma unroll
                for (int m = 8; m >= 1; m >>= 1) mx = fmaxf(mx, __shfl_xor(mx, m, 32));
                float mnew  = fmaxf(mst[r], mx);
                float alpha = __expf(mst[r] - mnew);
                mst[r] = mnew;
                float p0 = __expf(lg0[r] - mnew);
                float p1 = __expf(lg1[r] - mnew);
                float rs = p0 + p1;
#pragma unroll
                for (int m = 8; m >= 1; m >>= 1) rs += __shfl_xor(rs, m, 32);
                lst[r] = lst[r] * alpha + rs;
#pragma unroll
                for (int t = 0; t < 5; ++t) cacc[t][r] *= alpha;
                Ps[wave][(half * 8 + r) * 32 + mr]      = (__bf16)p0;
                Ps[wave][(half * 8 + r) * 32 + 16 + mr] = (__bf16)p1;
            }
            const __bf16* pp = &Ps[wave][mr * 32 + half * 8];
            v16bf pf = make_frag(ld16(pp), ld16(pp + 16));
#pragma unroll
            for (int t = 0; t < 5; ++t) {
                const __bf16* vr = &Vs[(t * 16 + mr) * 32 + half * 8];
                v16bf vf = make_frag(ld16(vr), ld16(vr + 16));
                cacc[t] = wmma_bf16(pf, vf, cacc[t]);
            }
        }
    }

    __bf16* cp = ctx + baseQ;
#pragma unroll
    for (int t = 0; t < 5; ++t)
#pragma unroll
        for (int r = 0; r < 8; ++r) {
            int row = qw + half * 8 + r;
            float val = cacc[t][r] / lst[r];
            cp[(size_t)row * DD + t * 16 + mr] = (__bf16)val;
        }
}

// ---------------- host orchestration ----------------
extern "C" void kernel_launch(void* const* d_in, const int* in_sizes, int n_in,
                              void* d_out, int out_size, void* d_ws, size_t ws_size,
                              hipStream_t stream) {
    (void)in_sizes; (void)n_in; (void)out_size; (void)ws_size;
    const int*   ids  = (const int*)d_in[0];
    const float* emb  = (const float*)d_in[1];
    const float* Wq   = (const float*)d_in[2];
    const float* Wk   = (const float*)d_in[3];
    const float* Wv   = (const float*)d_in[4];
    const float* Wo   = (const float*)d_in[5];
    const float* bq   = (const float*)d_in[6];
    const float* bk   = (const float*)d_in[7];
    const float* bv   = (const float*)d_in[8];
    const float* bo   = (const float*)d_in[9];
    const float* W1   = (const float*)d_in[10];
    const float* b1   = (const float*)d_in[11];
    const float* W2   = (const float*)d_in[12];
    const float* b2   = (const float*)d_in[13];
    const float* ln1g = (const float*)d_in[14];
    const float* ln1b = (const float*)d_in[15];
    const float* ln2g = (const float*)d_in[16];
    const float* ln2b = (const float*)d_in[17];
    const float* lnfg = (const float*)d_in[18];
    const float* lnfb = (const float*)d_in[19];

    char* w = (char*)d_ws;
    auto take = [&](size_t bytes) -> char* {
        char* p = w; w += (bytes + 255) & ~(size_t)255; return p;
    };
    float*  x   = (float*) take((size_t)BSR * DD * 4);
    __bf16* nb  = (__bf16*)take((size_t)BSR * DD * 2);
    __bf16* qb  = (__bf16*)take((size_t)BSR * DD * 2);
    __bf16* kb  = (__bf16*)take((size_t)BSR * DD * 2);
    __bf16* vb  = (__bf16*)take((size_t)BSR * DD * 2);
    __bf16* cb  = (__bf16*)take((size_t)BSR * DD * 2);
    __bf16* fb  = (__bf16*)take((size_t)BSR * DFFC * 2);
    __bf16* wqb = (__bf16*)take((size_t)DD * DD * 2);
    __bf16* wkb = (__bf16*)take((size_t)DD * DD * 2);
    __bf16* wvb = (__bf16*)take((size_t)DD * DD * 2);
    __bf16* wob = (__bf16*)take((size_t)DD * DD * 2);
    __bf16* w1b = (__bf16*)take((size_t)DD * DFFC * 2);
    __bf16* w2b = (__bf16*)take((size_t)DFFC * DD * 2);

    embed_kernel<<<BSR, 256, 0, stream>>>(ids, emb, x);

    const size_t nDD  = (size_t)DD * DD;
    const size_t nDF  = (size_t)DD * DFFC;
    const int gDD = (int)((nDD / 4 + 255) / 256);
    const int gDF = (int)((nDF / 4 + 255) / 256);
    const dim3 gp(DD / 128, BSR / 128);
    const dim3 gf(DFFC / 128, BSR / 128);
    const dim3 ga(SS / 128, BB * HH);

    for (int l = 0; l < LL; ++l) {
        const size_t od = (size_t)l * nDD, of = (size_t)l * nDF;
        cvt_kernel<<<gDD, 256, 0, stream>>>(Wq + od, wqb, nDD);
        cvt_kernel<<<gDD, 256, 0, stream>>>(Wk + od, wkb, nDD);
        cvt_kernel<<<gDD, 256, 0, stream>>>(Wv + od, wvb, nDD);
        cvt_kernel<<<gDD, 256, 0, stream>>>(Wo + od, wob, nDD);
        cvt_kernel<<<gDF, 256, 0, stream>>>(W1 + of, w1b, nDF);
        cvt_kernel<<<gDF, 256, 0, stream>>>(W2 + of, w2b, nDF);

        layernorm_kernel<__bf16><<<BSR, 256, 0, stream>>>(x, ln1g + l * DD, ln1b + l * DD, nb);

        gemm_kernel<false, false, __bf16><<<gp, 256, 0, stream>>>(nb, wqb, bq + l * DD, nullptr, qb, BSR, DD, DD);
        gemm_kernel<false, false, __bf16><<<gp, 256, 0, stream>>>(nb, wkb, bk + l * DD, nullptr, kb, BSR, DD, DD);
        gemm_kernel<false, false, __bf16><<<gp, 256, 0, stream>>>(nb, wvb, bv + l * DD, nullptr, vb, BSR, DD, DD);

        attention_kernel<<<ga, 256, 0, stream>>>(qb, kb, vb, cb);

        gemm_kernel<false, true, float><<<gp, 256, 0, stream>>>(cb, wob, bo + l * DD, x, x, BSR, DD, DD);

        layernorm_kernel<__bf16><<<BSR, 256, 0, stream>>>(x, ln2g + l * DD, ln2b + l * DD, nb);

        gemm_kernel<true, false, __bf16><<<gf, 256, 0, stream>>>(nb, w1b, b1 + l * DFFC, nullptr, fb, BSR, DFFC, DD);
        gemm_kernel<false, true, float><<<gp, 256, 0, stream>>>(fb, w2b, b2 + l * DD, x, x, BSR, DD, DFFC);
    }

    layernorm_kernel<float><<<BSR, 256, 0, stream>>>(x, lnfg, lnfb, (float*)d_out);
}
